// SplatAttentionPooling_80229989089751
// MI455X (gfx1250) — matrix operand order repaired and checked by reading, not compiled, not run on hardware
//
#include <hip/hip_runtime.h>

#define NROWS 8192
#define NCH   512
#define SLD   520  // padded LDS row stride in halves (1040B -> conflict-free)

typedef _Float16 v16h __attribute__((ext_vector_type(16)));
typedef _Float16 h8   __attribute__((ext_vector_type(8)));
typedef float    v8f  __attribute__((ext_vector_type(8)));
typedef unsigned int v4u __attribute__((ext_vector_type(4)));
typedef int          v4i __attribute__((ext_vector_type(4)));
typedef int          v8i __attribute__((ext_vector_type(8)));

#define WMMA_F16(Af, Bf, Cacc) \
  __builtin_amdgcn_wmma_f32_16x16x32_f16(false, (Af), false, (Bf), (short)0, (Cacc), false, false)

__device__ __forceinline__ float lane_xor_f(float v, int msk) {
  int lane = (int)(threadIdx.x & 31u);
  int idx  = ((lane ^ msk) << 2);
  return __int_as_float(__builtin_amdgcn_ds_bpermute(idx, __float_as_int(v)));
}

// A-operand fragment of a 16x32 f16 tile from a 16-row panel (row-major, stride LD).
// lane holds row (lane&15); elems 0..7 -> K=kh+0..7, elems 8..15 -> K=16+kh+0..7, kh=(lane&16)?8:0.
template <int LD>
__device__ __forceinline__ v16h load_fragA(const _Float16* base, int lane, int kc) {
  int r  = lane & 15;
  int kh = (lane & 16) >> 1;
  const _Float16* p = base + r * LD + kc + kh;
  h8 lo = *(const h8*)(p);
  h8 hi = *(const h8*)(p + 16);
  return __builtin_shufflevector(lo, hi, 0,1,2,3,4,5,6,7,8,9,10,11,12,13,14,15);
}

// B-operand fragment of a 32x16 f16 tile where B[k][n] = panel_row_n[k].
// lane holds col n=(lane&15); elems 0..7 -> K=kh+0..7, elems 8..15 -> K=16+kh+0..7, kh=(lane&16)?8:0.
template <int LD>
__device__ __forceinline__ v16h load_fragB(const _Float16* base, int lane, int kc) {
  int n  = lane & 15;
  int kh = (lane & 16) >> 1;
  const _Float16* p = base + n * LD + kc + kh;
  h8 lo = *(const h8*)(p);
  h8 hi = *(const h8*)(p + 16);
  return __builtin_shufflevector(lo, hi, 0,1,2,3,4,5,6,7,8,9,10,11,12,13,14,15);
}

// Tensor Data Mover: DMA one 16-row x 512-col f16 panel (row-major, stride 512)
// from global into LDS, with 16B padding after each 1024B row (matches SLD=520).
// D# per CDNA5 ISA ch.8: group0 {count, lds_addr, global_addr, type=2},
// group1 {data_size=2B, pad 7/3, tensor_dim0=512, tensor_dim1=8192,
//         tile_dim0=512, tile_dim1=16, tensor_dim0_stride=512}.
__device__ __forceinline__ void tdm_load_panel(const _Float16* gsrc_row0, void* lds_dst) {
  unsigned long long ga = (unsigned long long)(const void*)gsrc_row0;
  unsigned int ldsoff   = (unsigned int)(unsigned long long)lds_dst;  // low 32 = LDS offset
  v4u g0 = { 1u,                                   // count=1, is_restore=0, gather off
             ldsoff,                               // lds_addr
             (unsigned int)(ga & 0xFFFFFFFFull),   // global_addr[31:0]
             (unsigned int)(((ga >> 32) & 0x01FFFFFFull) | 0x80000000u) };  // addr[56:32]|type=2
  v8i g1 = { (int)0x07D10000,        // data_size=1(2B), pad_enable, pad_interval=7, pad_amount=3
             (int)(512u << 16),      // abar=0 | tensor_dim0[15:0]=512
             (int)0x20000000,        // tensor_dim0[31:16]=0 | tensor_dim1[15:0]=8192
             (int)0x02000000,        // tensor_dim1[31:16]=0 | tile_dim0=512
             16,                     // tile_dim1=16 | tile_dim2=0
             512,                    // tensor_dim0_stride[31:0]=512
             0, 0 };                 // stride0 hi / tensor_dim1_stride = 0
  v4i z4 = { 0, 0, 0, 0 };
#if __clang_major__ >= 23
  v8i z8 = { 0, 0, 0, 0, 0, 0, 0, 0 };
  __builtin_amdgcn_tensor_load_to_lds(g0, g1, z4, z4, z8, 0);
#else
  __builtin_amdgcn_tensor_load_to_lds(g0, g1, z4, z4, 0);
#endif
}

__global__ void cvt_kernel(const float* __restrict__ x, _Float16* __restrict__ xh) {
  int i  = (int)(blockIdx.x * 256 + threadIdx.x);
  xh[i] = (_Float16)x[i];
}

// Pass 1: per-row online softmax stats over a half of the key range.
__global__ __launch_bounds__(128) void pass1_kernel(const _Float16* __restrict__ xh,
                                                    float* __restrict__ m_part,
                                                    float* __restrict__ l_part) {
  __shared__ _Float16 sP[2][16 * SLD];
  const int lane = (int)(threadIdx.x & 31u);
  const int wave = (int)(threadIdx.x >> 5);
  const int i0   = ((int)blockIdx.x * 4 + wave) * 16;
  const int jbeg = (int)blockIdx.y * (NROWS / 2);
  const int ntiles = (NROWS / 2) / 16;

  v16h A[16];
#pragma unroll
  for (int k = 0; k < 16; ++k) A[k] = load_fragA<NCH>(xh + (size_t)i0 * NCH, lane, k * 32);

  float m[8], l[8];
#pragma unroll
  for (int v = 0; v < 8; ++v) { m[v] = -3.0e38f; l[v] = 0.0f; }

  if (wave == 0) tdm_load_panel(xh + (size_t)jbeg * NCH, &sP[0][0]);

  for (int t = 0; t < ntiles; ++t) {
    if (wave == 0) __builtin_amdgcn_s_wait_tensorcnt(0);
    __syncthreads();  // panel t visible; buffer (t+1)&1 no longer read by anyone
    if (wave == 0 && t + 1 < ntiles)
      tdm_load_panel(xh + (size_t)(jbeg + (t + 1) * 16) * NCH, &sP[(t + 1) & 1][0]);
    const _Float16* sp = &sP[t & 1][0];

    v8f acc0 = {}, acc1 = {}, acc2 = {}, acc3 = {};
#pragma unroll
    for (int k = 0; k < 16; k += 4) {
      acc0 = WMMA_F16(A[k + 0], load_fragB<SLD>(sp, lane, (k + 0) * 32), acc0);
      acc1 = WMMA_F16(A[k + 1], load_fragB<SLD>(sp, lane, (k + 1) * 32), acc1);
      acc2 = WMMA_F16(A[k + 2], load_fragB<SLD>(sp, lane, (k + 2) * 32), acc2);
      acc3 = WMMA_F16(A[k + 3], load_fragB<SLD>(sp, lane, (k + 3) * 32), acc3);
    }
#pragma unroll
    for (int v = 0; v < 8; ++v) {
      float s = acc0[v] + acc1[v] + acc2[v] + acc3[v];
      float d = s - m[v];
      float e = __expf(-fabsf(d));
      if (d > 0.0f) { l[v] = l[v] * e + 1.0f; m[v] = s; } else { l[v] += e; }
    }
  }

  // merge stats across the 16 lanes of each half (columns seen by this lane group)
#pragma unroll
  for (int v = 0; v < 8; ++v) {
#pragma unroll
    for (int msk = 1; msk <= 8; msk <<= 1) {
      float mo = lane_xor_f(m[v], msk);
      float lo = lane_xor_f(l[v], msk);
      float nm = fmaxf(m[v], mo);
      l[v] = l[v] * __expf(m[v] - nm) + lo * __expf(mo - nm);
      m[v] = nm;
    }
  }
  if ((lane & 15) == 0) {
    int rbase = i0 + ((lane & 16) >> 1);
    int off   = (int)blockIdx.y * NROWS;
#pragma unroll
    for (int v = 0; v < 8; ++v) {
      m_part[off + rbase + v] = m[v];
      l_part[off + rbase + v] = l[v];
    }
  }
}

__global__ void merge_kernel(const float* __restrict__ mp, const float* __restrict__ lp,
                             float* __restrict__ m_g, float* __restrict__ rl_g) {
  int i   = (int)(blockIdx.x * 256 + threadIdx.x);
  float m0 = mp[i], m1 = mp[NROWS + i];
  float l0 = lp[i], l1 = lp[NROWS + i];
  float mm = fmaxf(m0, m1);
  float l  = l0 * __expf(m0 - mm) + l1 * __expf(m1 - mm);
  m_g[i]  = mm;
  rl_g[i] = 1.0f / l;
}

// Pass 2: w_j = sum_i exp(s_ij - m_i) / l_i   (each wave owns 16 key columns)
__global__ __launch_bounds__(128) void pass2_kernel(const _Float16* __restrict__ xh,
                                                    const float* __restrict__ m_g,
                                                    const float* __restrict__ rl_g,
                                                    float* __restrict__ w_part) {
  __shared__ _Float16 sP[2][16 * SLD];
  const int lane = (int)(threadIdx.x & 31u);
  const int wave = (int)(threadIdx.x >> 5);
  const int j0   = ((int)blockIdx.x * 4 + wave) * 16;
  const int ibeg = (int)blockIdx.y * (NROWS / 2);
  const int ntiles = (NROWS / 2) / 16;
  const int h8o  = (lane & 16) >> 1;

  v16h B[16];
#pragma unroll
  for (int k = 0; k < 16; ++k) B[k] = load_fragB<NCH>(xh + (size_t)j0 * NCH, lane, k * 32);

  if (wave == 0) tdm_load_panel(xh + (size_t)ibeg * NCH, &sP[0][0]);

  float wacc = 0.0f;
  for (int t = 0; t < ntiles; ++t) {
    int i0 = ibeg + t * 16;
    if (wave == 0) __builtin_amdgcn_s_wait_tensorcnt(0);
    __syncthreads();
    if (wave == 0 && t + 1 < ntiles)
      tdm_load_panel(xh + (size_t)(i0 + 16) * NCH, &sP[(t + 1) & 1][0]);
    const _Float16* sp = &sP[t & 1][0];

    v8f acc0 = {}, acc1 = {}, acc2 = {}, acc3 = {};
#pragma unroll
    for (int k = 0; k < 16; k += 4) {
      acc0 = WMMA_F16(load_fragA<SLD>(sp, lane, (k + 0) * 32), B[k + 0], acc0);
      acc1 = WMMA_F16(load_fragA<SLD>(sp, lane, (k + 1) * 32), B[k + 1], acc1);
      acc2 = WMMA_F16(load_fragA<SLD>(sp, lane, (k + 2) * 32), B[k + 2], acc2);
      acc3 = WMMA_F16(load_fragA<SLD>(sp, lane, (k + 3) * 32), B[k + 3], acc3);
    }
#pragma unroll
    for (int v = 0; v < 8; ++v) {
      float s = acc0[v] + acc1[v] + acc2[v] + acc3[v];
      int   r = i0 + h8o + v;
      wacc += __expf(s - m_g[r]) * rl_g[r];
    }
  }
  wacc += lane_xor_f(wacc, 16);  // fold the two row-halves (same column)
  if (lane < 16) w_part[(int)blockIdx.y * NROWS + j0 + lane] = wacc;
}

// Pass 3: y = (1/N) * sum_j w_j * x_j  (fp32), then L2-normalize.
__global__ __launch_bounds__(512) void pass3_kernel(const float* __restrict__ x,
                                                    const float* __restrict__ w_part,
                                                    float* __restrict__ out) {
  __shared__ float red[NCH];
  int c = (int)threadIdx.x;
  float sum = 0.0f;
#pragma unroll 4
  for (int j = 0; j < NROWS; ++j) {
    float wj = w_part[j] + w_part[NROWS + j];
    sum += wj * x[(size_t)j * NCH + c];
  }
  float y = sum * (1.0f / (float)NROWS);
  red[c] = y * y;
  __syncthreads();
  for (int s = 256; s > 0; s >>= 1) {
    if (c < s) red[c] += red[c + s];
    __syncthreads();
  }
  float norm = fmaxf(sqrtf(red[0]), 1e-12f);
  out[c] = y / norm;
}

extern "C" void kernel_launch(void* const* d_in, const int* in_sizes, int n_in,
                              void* d_out, int out_size, void* d_ws, size_t ws_size,
                              hipStream_t stream) {
  (void)in_sizes; (void)n_in; (void)out_size; (void)ws_size;
  const float* x = (const float*)d_in[0];
  float* out = (float*)d_out;

  char* ws = (char*)d_ws;
  size_t off = 0;
  _Float16* xh = (_Float16*)(ws + off); off += (size_t)NROWS * NCH * sizeof(_Float16); // 8 MB
  float* m_part = (float*)(ws + off);   off += (size_t)2 * NROWS * sizeof(float);
  float* l_part = (float*)(ws + off);   off += (size_t)2 * NROWS * sizeof(float);
  float* m_g    = (float*)(ws + off);   off += (size_t)NROWS * sizeof(float);
  float* rl_g   = (float*)(ws + off);   off += (size_t)NROWS * sizeof(float);
  float* w_part = (float*)(ws + off);   off += (size_t)2 * NROWS * sizeof(float);

  cvt_kernel<<<(NROWS * NCH) / 256, 256, 0, stream>>>(x, xh);
  pass1_kernel<<<dim3(NROWS / 64, 2), 128, 0, stream>>>(xh, m_part, l_part);
  merge_kernel<<<NROWS / 256, 256, 0, stream>>>(m_part, l_part, m_g, rl_g);
  pass2_kernel<<<dim3(NROWS / 64, 2), 128, 0, stream>>>(xh, m_g, rl_g, w_part);
  pass3_kernel<<<1, NCH, 0, stream>>>(x, w_part, out);
}
